// Encoder_33157147525485
// MI455X (gfx1250) — compile-verified, matrix operand order checked by reading, not run on hardware
//
#include <hip/hip_runtime.h>
#include <hip/hip_bf16.h>

#define BB 8
#define SS 1024
#define DD 512
#define HH 8
#define DHH 64
#define MLP 256
#define MM (BB * SS)   // 8192 rows
#define LN_EPS 1e-5f

typedef __attribute__((ext_vector_type(16))) __bf16 bf16x16;
typedef __attribute__((ext_vector_type(8)))  float  floatx8;

union FragBF16 {
    bf16x16 v;
    uint4   q[2];
    unsigned int u[8];
    unsigned short h[16];
};

__device__ __forceinline__ unsigned short f2bf(float f) {
    unsigned int u = __float_as_uint(f);
    unsigned int r = u + 0x7FFFu + ((u >> 16) & 1u);   // round-to-nearest-even
    return (unsigned short)(r >> 16);
}

__device__ __forceinline__ floatx8 zero8() {
    floatx8 z;
#pragma unroll
    for (int i = 0; i < 8; ++i) z[i] = 0.0f;
    return z;
}

// Load a 16x32 bf16 fragment (A-layout / B-layout identical per lane):
// lane L (l16 = L&15, half = L>>4) owns row l16; VGPR0..3 hold K = half*8..half*8+7,
// VGPR4..7 hold K = 16+half*8..16+half*8+7 -> two contiguous 16-byte loads.
__device__ __forceinline__ void load_frag(FragBF16& f, const unsigned short* rowk0, int half) {
    const unsigned short* p = rowk0 + half * 8;
    f.q[0] = *reinterpret_cast<const uint4*>(p);
    f.q[1] = *reinterpret_cast<const uint4*>(p + 16);
}

__device__ __forceinline__ floatx8 wmma_bf16(const FragBF16& a, const FragBF16& b, floatx8 c) {
    return __builtin_amdgcn_wmma_f32_16x16x32_bf16(false, a.v, false, b.v, (short)0, c, false, false);
}

__device__ __forceinline__ float gelu_exact(float x) {
    return 0.5f * x * (1.0f + erff(x * 0.70710678118654752f));
}

// ---------------------------------------------------------------------------
// Elementwise: x = x + emb (broadcast over batch); write fp32 + bf16 copies
// ---------------------------------------------------------------------------
__global__ void add_emb_kernel(const float* __restrict__ x, const float* __restrict__ emb,
                               float* __restrict__ xf, unsigned short* __restrict__ xb) {
    int idx = blockIdx.x * blockDim.x + threadIdx.x;
    if (idx >= MM * DD) return;
    float v = x[idx] + emb[idx & (SS * DD - 1)];
    xf[idx] = v;
    xb[idx] = f2bf(v);
}

// ---------------------------------------------------------------------------
// Weight prep: W [K,N] fp32 row-major  ->  WT [N,K] bf16 (contiguous K rows)
// ---------------------------------------------------------------------------
__global__ void prep_wt_kernel(const float* __restrict__ W, unsigned short* __restrict__ WT,
                               int K, int N) {
    int idx = blockIdx.x * blockDim.x + threadIdx.x;
    if (idx >= K * N) return;
    int n = idx / K, k = idx - n * K;
    WT[idx] = f2bf(W[k * N + n]);
}

// ---------------------------------------------------------------------------
// GEMM: out[M,N] = A_bf16[M,K] * WT_bf16[N,K]^T + bias, fp32 accumulate.
// One wave computes a 16x64 strip (4 WMMA accumulators, A-fragment reused 4x).
// MODE: 0 = fp32 out, 1 = bf16 out, 2 = gelu + bf16 out, 3 = bf16 out
//       transposed per-head as VT[b,h,dh,s]
// ---------------------------------------------------------------------------
template <int MODE>
__global__ void gemm_wmma_kernel(const unsigned short* __restrict__ A,
                                 const unsigned short* __restrict__ WT,
                                 const float* __restrict__ bias,
                                 float* __restrict__ outF,
                                 unsigned short* __restrict__ outB,
                                 int M, int N, int K) {
    const int lane = threadIdx.x & 31;
    const int wave = threadIdx.x >> 5;
    const int l16  = lane & 15;
    const int half = lane >> 4;

    const int nSup = N >> 6;
    const int tile = blockIdx.x * (blockDim.x >> 5) + wave;
    const int mT = tile / nSup;
    const int nS = tile - mT * nSup;
    if (mT * 16 >= M) return;
    const int mBase = mT * 16;
    const int nBase = nS * 64;

    floatx8 acc[4];
#pragma unroll
    for (int t = 0; t < 4; ++t) acc[t] = zero8();

    const unsigned short* aRow = A + (size_t)(mBase + l16) * K;
    for (int k0 = 0; k0 < K; k0 += 32) {
        FragBF16 af;
        load_frag(af, aRow + k0, half);
#pragma unroll
        for (int t = 0; t < 4; ++t) {
            FragBF16 bf;
            load_frag(bf, WT + (size_t)(nBase + t * 16 + l16) * K + k0, half);
            acc[t] = wmma_bf16(af, bf, acc[t]);
        }
    }

#pragma unroll
    for (int t = 0; t < 4; ++t) {
        const int n = nBase + t * 16 + l16;
        const float bv = bias[n];
#pragma unroll
        for (int v = 0; v < 8; ++v) {
            const int m = mBase + v + half * 8;
            float val = acc[t][v] + bv;
            if (MODE == 2) val = gelu_exact(val);
            if (MODE == 0) {
                outF[(size_t)m * N + n] = val;
            } else if (MODE == 3) {
                const int bi = m >> 10;         // m / SS
                const int si = m & (SS - 1);
                const int hh = n >> 6;
                const int dh = n & 63;
                outB[((size_t)((bi * HH + hh) * DHH + dh)) * SS + si] = f2bf(val);
            } else {
                outB[(size_t)m * N + n] = f2bf(val);
            }
        }
    }
}

// ---------------------------------------------------------------------------
// Flash attention, one wave per 16-query tile of one (b,h).
// scores = Q K^T / sqrt(S); softmax over full row; mask att where score <= 0
// (sigmoid(s) > 0.5  <=>  s > 0), no renormalization after masking.
//
// Layout trick: raw fp32 scores go through LDS (C-layout store, A-layout
// load), so each lane owns one score row (row = l16, its `half` selecting a
// complementary 16 of the 32 keys). Softmax stats (m, l, alpha) are per-lane
// SCALARS; row reductions are lane-local + a single shfl_xor(16). The masked
// exp(P) values are produced directly in A-fragment order (packed bf16 in
// registers, no second LDS trip). Only alpha (per iter) and l (once) need
// broadcasting back into the C-layout accumulators (8 shfl each).
// ---------------------------------------------------------------------------
__global__ void __launch_bounds__(128)
flash_attn_kernel(const unsigned short* __restrict__ Qb,
                  const unsigned short* __restrict__ Kb,
                  const unsigned short* __restrict__ VT,
                  unsigned short* __restrict__ Ob) {
    __shared__ alignas(16) float ldsS[4][16 * 32];

    const int lane = threadIdx.x & 31;
    const int wave = threadIdx.x >> 5;
    const int l16  = lane & 15;
    const int half = lane >> 4;

    const int id = blockIdx.x * 4 + wave;          // B*H*(S/16) = 4096 waves total
    const int qt = id & 63;
    const int bh = id >> 6;
    const int h  = bh & (HH - 1);
    const int b  = bh >> 3;
    const int qBase = qt * 16;

    // Q fragments (dh 0..31 and 32..63), rows contiguous in [B,S,D] bf16
    const unsigned short* qRow = Qb + (size_t)(b * SS + qBase + l16) * DD + h * DHH;
    FragBF16 qf0, qf1;
    load_frag(qf0, qRow, half);
    load_frag(qf1, qRow + 32, half);

    float mrun = -1e30f;    // running row max  (row = l16)
    float lrun = 0.0f;      // running row denominator
    floatx8 acc[4];
#pragma unroll
    for (int t = 0; t < 4; ++t) acc[t] = zero8();

    const unsigned short* vtBase = VT + (size_t)((b * HH + h) * DHH) * SS;
    float* myRow = &ldsS[wave][0];                 // C-layout store base
    const float4* aRowLds =
        reinterpret_cast<const float4*>(&ldsS[wave][l16 * 32 + half * 8]);

    for (int kt = 0; kt < SS / 32; ++kt) {
        const int kBase = kt * 32;

        // ---- scores: two 16x16 tiles over 32 keys; store raw fp32 C-layout ----
#pragma unroll
        for (int sub = 0; sub < 2; ++sub) {
            const unsigned short* kRow =
                Kb + (size_t)(b * SS + kBase + sub * 16 + l16) * DD + h * DHH;
            FragBF16 kf0, kf1;
            load_frag(kf0, kRow, half);
            load_frag(kf1, kRow + 32, half);
            floatx8 sc = zero8();
            sc = wmma_bf16(qf0, kf0, sc);
            sc = wmma_bf16(qf1, kf1, sc);
#pragma unroll
            for (int v = 0; v < 8; ++v)
                myRow[(v + half * 8) * 32 + sub * 16 + l16] = sc[v] * 0.03125f;
        }

        // C-layout -> A-layout through LDS (LDS ops are in-order per wave)
        asm volatile("s_wait_dscnt 0x0" ::: "memory");
        float sv[16];
        {
            float4 r0 = aRowLds[0];                 // k = half*8 .. half*8+7
            float4 r1 = aRowLds[1];
            float4 r2 = aRowLds[4];                 // k = 16+half*8 .. 16+half*8+7
            float4 r3 = aRowLds[5];
            sv[0]=r0.x; sv[1]=r0.y; sv[2]=r0.z; sv[3]=r0.w;
            sv[4]=r1.x; sv[5]=r1.y; sv[6]=r1.z; sv[7]=r1.w;
            sv[8]=r2.x; sv[9]=r2.y; sv[10]=r2.z; sv[11]=r2.w;
            sv[12]=r3.x; sv[13]=r3.y; sv[14]=r3.z; sv[15]=r3.w;
        }

        // ---- per-lane softmax stats (row = l16) ----
        float lmax = sv[0];
#pragma unroll
        for (int i = 1; i < 16; ++i) lmax = fmaxf(lmax, sv[i]);
        lmax = fmaxf(lmax, __shfl_xor(lmax, 16, 32));   // combine k-halves
        const float mnew  = fmaxf(mrun, lmax);
        const float alpha = __expf(mrun - mnew);
        mrun = mnew;

        FragBF16 pf;
        float lsum = 0.0f;
#pragma unroll
        for (int i = 0; i < 16; ++i) {
            float p = __expf(sv[i] - mnew);
            lsum += p;
            pf.h[i] = f2bf((sv[i] > 0.0f) ? p : 0.0f);  // mask numerator only
        }
        lsum += __shfl_xor(lsum, 16, 32);
        lrun = lrun * alpha + lsum;

        // ---- rescale C-layout accumulators (alpha of row v + 8*half) ----
        float alphaC[8];
#pragma unroll
        for (int v = 0; v < 8; ++v)
            alphaC[v] = __shfl(alpha, v + 8 * half, 32);
#pragma unroll
        for (int t = 0; t < 4; ++t) {
#pragma unroll
            for (int v = 0; v < 8; ++v) acc[t][v] *= alphaC[v];
            FragBF16 vf;
            load_frag(vf, vtBase + (size_t)(t * 16 + l16) * SS + kBase, half);
            acc[t] = wmma_bf16(pf, vf, acc[t]);
        }
    }

    // ---- epilogue: out = acc / l (l of row v + 8*half), write bf16 O ----
    float lC[8];
#pragma unroll
    for (int v = 0; v < 8; ++v)
        lC[v] = __shfl(lrun, v + 8 * half, 32);
#pragma unroll
    for (int t = 0; t < 4; ++t) {
#pragma unroll
        for (int v = 0; v < 8; ++v) {
            const int m = qBase + v + half * 8;
            float val = acc[t][v] / lC[v];
            Ob[(size_t)(b * SS + m) * DD + h * DHH + t * 16 + l16] = f2bf(val);
        }
    }
}

// ---------------------------------------------------------------------------
// out = resid + LayerNorm(src) * g + b; one wave32 per 512-wide row.
// Writes fp32 (outF) and bf16 (outB) copies.
// ---------------------------------------------------------------------------
__global__ void ln_residual_kernel(const float* __restrict__ src,
                                   const float* __restrict__ resid,
                                   const float* __restrict__ g,
                                   const float* __restrict__ bb,
                                   float* __restrict__ outF,
                                   unsigned short* __restrict__ outB) {
    const int lane = threadIdx.x & 31;
    const int wave = threadIdx.x >> 5;
    const int row  = blockIdx.x * (blockDim.x >> 5) + wave;
    if (row >= MM) return;
    const float* x = src + (size_t)row * DD;

    float vals[16];
    float s = 0.0f, s2 = 0.0f;
#pragma unroll
    for (int i = 0; i < 16; ++i) {
        float v = x[lane + i * 32];
        vals[i] = v;
        s += v;
        s2 += v * v;
    }
#pragma unroll
    for (int off = 16; off >= 1; off >>= 1) {
        s  += __shfl_xor(s, off, 32);
        s2 += __shfl_xor(s2, off, 32);
    }
    const float mean = s * (1.0f / DD);
    const float var  = s2 * (1.0f / DD) - mean * mean;
    const float rstd = rsqrtf(var + LN_EPS);

#pragma unroll
    for (int i = 0; i < 16; ++i) {
        const int d = lane + i * 32;
        float y = (vals[i] - mean) * rstd * g[d] + bb[d];
        float o = resid[(size_t)row * DD + d] + y;
        outF[(size_t)row * DD + d] = o;
        outB[(size_t)row * DD + d] = f2bf(o);
    }
}

// ---------------------------------------------------------------------------
// Host orchestration
// ---------------------------------------------------------------------------
extern "C" void kernel_launch(void* const* d_in, const int* in_sizes, int n_in,
                              void* d_out, int out_size, void* d_ws, size_t ws_size,
                              hipStream_t stream) {
    (void)in_sizes; (void)n_in; (void)out_size; (void)ws_size;

    const float* x   = (const float*)d_in[0];
    const float* emb = (const float*)d_in[1];
    // params in _layer_params insertion order
    const float* const* p1 = (const float* const*)(&d_in[2]);
    const float* const* p2 = (const float* const*)(&d_in[16]);
    // indices: 0 Wq, 1 bq, 2 Wk, 3 bk, 4 Wv, 5 bv, 6 Wo, 7 bo,
    //          8 fc1_w, 9 fc1_b, 10 fc2_w, 11 fc2_b, 12 ln_g, 13 ln_b

    size_t off = 0;
    auto carve = [&](size_t bytes) -> void* {
        void* p = (char*)d_ws + off;
        off += (bytes + 255) & ~(size_t)255;
        return p;
    };
    float*          xf   = (float*)         carve((size_t)MM * DD * 4);
    unsigned short* xb   = (unsigned short*)carve((size_t)MM * DD * 2);
    float*          tmpf = (float*)         carve((size_t)MM * DD * 4);
    unsigned short* qb   = (unsigned short*)carve((size_t)MM * DD * 2);
    unsigned short* kb   = (unsigned short*)carve((size_t)MM * DD * 2);
    unsigned short* vtb  = (unsigned short*)carve((size_t)MM * DD * 2);
    unsigned short* ob   = (unsigned short*)carve((size_t)MM * DD * 2);
    unsigned short* hb   = (unsigned short*)carve((size_t)MM * MLP * 2);
    unsigned short* wtq  = (unsigned short*)carve((size_t)DD * DD * 2);
    unsigned short* wtk  = (unsigned short*)carve((size_t)DD * DD * 2);
    unsigned short* wtv  = (unsigned short*)carve((size_t)DD * DD * 2);
    unsigned short* wto  = (unsigned short*)carve((size_t)DD * DD * 2);
    unsigned short* wt1  = (unsigned short*)carve((size_t)DD * MLP * 2);
    unsigned short* wt2  = (unsigned short*)carve((size_t)MLP * DD * 2);

    // x = x + emb
    add_emb_kernel<<<(MM * DD + 255) / 256, 256, 0, stream>>>(x, emb, xf, xb);

    auto run_layer = [&](const float* const* p, float* final_out) {
        // weight prep (bf16, transposed to [N,K])
        prep_wt_kernel<<<(DD * DD  + 255) / 256, 256, 0, stream>>>(p[0],  wtq, DD,  DD);
        prep_wt_kernel<<<(DD * DD  + 255) / 256, 256, 0, stream>>>(p[2],  wtk, DD,  DD);
        prep_wt_kernel<<<(DD * DD  + 255) / 256, 256, 0, stream>>>(p[4],  wtv, DD,  DD);
        prep_wt_kernel<<<(DD * DD  + 255) / 256, 256, 0, stream>>>(p[6],  wto, DD,  DD);
        prep_wt_kernel<<<(DD * MLP + 255) / 256, 256, 0, stream>>>(p[8],  wt1, DD,  MLP);
        prep_wt_kernel<<<(MLP * DD + 255) / 256, 256, 0, stream>>>(p[10], wt2, MLP, DD);

        // Q, K (bf16), V (bf16, transposed per head)
        gemm_wmma_kernel<1><<<512, 256, 0, stream>>>(xb, wtq, p[1], nullptr, qb, MM, DD, DD);
        gemm_wmma_kernel<1><<<512, 256, 0, stream>>>(xb, wtk, p[3], nullptr, kb, MM, DD, DD);
        gemm_wmma_kernel<3><<<512, 256, 0, stream>>>(xb, wtv, p[5], nullptr, vtb, MM, DD, DD);

        // fused attention -> bf16 O
        flash_attn_kernel<<<1024, 128, 0, stream>>>(qb, kb, vtb, ob);

        // output projection -> fp32, then x = x + LN(attn_out)
        gemm_wmma_kernel<0><<<512, 256, 0, stream>>>(ob, wto, p[7], tmpf, nullptr, MM, DD, DD);
        ln_residual_kernel<<<1024, 256, 0, stream>>>(tmpf, xf, p[12], p[13], xf, xb);

        // MLP: gelu(x@fc1+b1) -> bf16, @fc2+b2 -> fp32, x = x + LN(mlp_out)
        gemm_wmma_kernel<2><<<256, 256, 0, stream>>>(xb, wt1, p[9],  nullptr, hb, MM, MLP, DD);
        gemm_wmma_kernel<0><<<512, 256, 0, stream>>>(hb, wt2, p[11], tmpf, nullptr, MM, DD, MLP);
        ln_residual_kernel<<<1024, 256, 0, stream>>>(tmpf, xf, p[12], p[13], final_out, xb);
    };

    run_layer(p1, xf);
    run_layer(p2, (float*)d_out);
}
